// MultiHeadAttention_11458972745856
// MI455X (gfx1250) — matrix operand order repaired and checked by reading, not compiled, not run on hardware
//
#include <hip/hip_runtime.h>
#include <hip/hip_bf16.h>
#include <math.h>

// MI455X / gfx1250, wave32. All matmuls via v_wmma_f32_16x16x32_f16.
// GEMMs: double-buffered LDS staging of the block-shared B tile via the
// CDNA5 async load-to-LDS path (ASYNCcnt), straight-line steady state.
// Attention: fused causal flash-attention, software-pipelined K/V loads.

typedef __attribute__((ext_vector_type(16))) _Float16 v16h;
typedef __attribute__((ext_vector_type(8)))  _Float16 v8h;
typedef __attribute__((ext_vector_type(4)))  _Float16 v4h;
typedef __attribute__((ext_vector_type(8)))  float    v8f;
typedef __attribute__((ext_vector_type(4)))  int      v4i;

#define SEQ      4096
#define D_MODEL  1024
#define NUM_HEADS 16
#define DK        64

#if __has_builtin(__builtin_amdgcn_global_load_async_to_lds_b128) && \
    __has_builtin(__builtin_amdgcn_s_wait_asynccnt)
#define USE_ASYNC_LDS 1
typedef __attribute__((address_space(1))) v4i gas_v4i;   // global int4
typedef __attribute__((address_space(3))) v4i las_v4i;   // LDS int4
#else
#define USE_ASYNC_LDS 0
#endif

static __device__ __forceinline__ v8f wmma_f16(v16h a, v16h b, v8f c) {
  // (neg_a, A, neg_b, B, c_mod, C, reuse_a, reuse_b)
  return __builtin_amdgcn_wmma_f32_16x16x32_f16(false, a, false, b, (short)0, c,
                                                false, false);
}

// A-tile fragment (16x32 f16): lane lo holds row M=lo, hi selects contiguous K-half (16 elems)
static __device__ __forceinline__ v16h load_afrag(const _Float16* p) {
  v16h a;
  ((v8h*)&a)[0] = *(const v8h*)p;
  ((v8h*)&a)[1] = *(const v8h*)(p + 8);
  return a;
}
// B-tile fragment (32x16 f16) from N-major storage: 32 contiguous bytes/lane
static __device__ __forceinline__ v16h load_bfrag(const _Float16* p) {
  v16h b;
  ((v8h*)&b)[0] = *(const v8h*)p;
  ((v8h*)&b)[1] = *(const v8h*)(p + 8);
  return b;
}

static __device__ __forceinline__ float rowmax16(float v) {
  v = fmaxf(v, __shfl_xor(v, 1, 32));
  v = fmaxf(v, __shfl_xor(v, 2, 32));
  v = fmaxf(v, __shfl_xor(v, 4, 32));
  v = fmaxf(v, __shfl_xor(v, 8, 32));
  return v;
}
static __device__ __forceinline__ float rowsum16(float v) {
  v += __shfl_xor(v, 1, 32);
  v += __shfl_xor(v, 2, 32);
  v += __shfl_xor(v, 4, 32);
  v += __shfl_xor(v, 8, 32);
  return v;
}

// 32 bytes global -> LDS, async (ASYNCcnt) when available
static __device__ __forceinline__ void stage32(_Float16* dstLds,
                                               const _Float16* srcGlob) {
#if USE_ASYNC_LDS
  __builtin_amdgcn_global_load_async_to_lds_b128(
      (gas_v4i*)srcGlob, (las_v4i*)dstLds, 0, 0);
  __builtin_amdgcn_global_load_async_to_lds_b128(
      (gas_v4i*)(srcGlob + 8), (las_v4i*)(dstLds + 8), 0, 0);
#else
  *(v8h*)dstLds       = *(const v8h*)srcGlob;
  *(v8h*)(dstLds + 8) = *(const v8h*)(srcGlob + 8);
#endif
}

// ---------------------------------------------------------------- f32 -> f16
__global__ void __launch_bounds__(256) cvt_f32_f16_kernel(
    const float* __restrict__ in, _Float16* __restrict__ out, int n) {
  int i = (blockIdx.x * 256 + threadIdx.x) * 4;
  if (i < n) {
    float4 f = *(const float4*)(in + i);
    v4h h;
    h[0] = (_Float16)f.x; h[1] = (_Float16)f.y;
    h[2] = (_Float16)f.z; h[3] = (_Float16)f.w;
    *(v4h*)(out + i) = h;
  }
}

// -------------------------------------------------------------- WMMA GEMM
// out = scale * (A[M][K] @ W[N][K]^T + bias). W (nn.Linear layout, N-major)
// is the natural CDNA5 WMMA-B layout. The 64x32 B tile shared by all 4 waves
// of the block is staged in LDS (double-buffered, async loads).
// mode 0: f16 out[M][N];  mode 1: f16 outT[N][M] (for V);  mode 2: f32 out[M][N]
__global__ void __launch_bounds__(128) gemm_wmma_kernel(
    const _Float16* __restrict__ A, const _Float16* __restrict__ W,
    const float* __restrict__ bias, void* __restrict__ out,
    int M, int N, int K, float scale, int mode) {
  __shared__ _Float16 Bt[2][64 * 40];      // 64 rows x 32 halves, stride 40
  const int tid  = threadIdx.x;
  const int lane = tid & 31;
  const int wid  = tid >> 5;
  const int lo = lane & 15, hi = lane >> 4;
  const int m0 = blockIdx.x * 64 + wid * 16;   // 4 waves stacked in M
  const int n0 = blockIdx.y * 64;              // each wave: 16 x 64 tile

  // cooperative staging: thread t covers B row j = t/2, 32-byte half seg = t&1
  const int j = tid >> 1, seg = tid & 1;
  const _Float16* wsrc = W + (size_t)(n0 + j) * K + seg * 16;
  _Float16* wl[2] = { &Bt[0][j * 40 + seg * 16], &Bt[1][j * 40 + seg * 16] };

  v8f acc[4] = {};
  const _Float16* arow = A + (size_t)(m0 + lo) * K + hi * 16;

  stage32(wl[0], wsrc);                        // prime buffer 0 (kc = 0)
  const int NK = K / 32;

  // steady state: straight-line stage(i+1) -> wait(2) -> barrier -> 4 WMMAs
  for (int i = 0; i < NK - 1; ++i) {
    const int kc = i * 32;
    stage32(wl[(i + 1) & 1], wsrc + kc + 32);
    __builtin_prefetch(arow + kc + 128, 0, 3);
    v16h a = load_afrag(arow + kc);
#if USE_ASYNC_LDS
    __builtin_amdgcn_s_wait_asynccnt(2);       // stage i landed (i+1 in flight)
#endif
    __syncthreads();                           // all waves' stage(i) visible
    const _Float16* bbase = &Bt[i & 1][0];
    v16h b[4];
#pragma unroll
    for (int nt = 0; nt < 4; ++nt) {           // batch LDS reads, then WMMAs
      const _Float16* brow = bbase + (nt * 16 + lo) * 40 + hi * 16;
      ((v8h*)&b[nt])[0] = *(const v8h*)brow;
      ((v8h*)&b[nt])[1] = *(const v8h*)(brow + 8);
    }
#pragma unroll
    for (int nt = 0; nt < 4; ++nt) acc[nt] = wmma_f16(a, b[nt], acc[nt]);
    __syncthreads();                           // reads done before overwrite
  }
  {                                            // epilogue iteration (no stage)
    const int kc = (NK - 1) * 32;
    v16h a = load_afrag(arow + kc);
#if USE_ASYNC_LDS
    __builtin_amdgcn_s_wait_asynccnt(0);
#endif
    __syncthreads();
    const _Float16* bbase = &Bt[(NK - 1) & 1][0];
    v16h b[4];
#pragma unroll
    for (int nt = 0; nt < 4; ++nt) {
      const _Float16* brow = bbase + (nt * 16 + lo) * 40 + hi * 16;
      ((v8h*)&b[nt])[0] = *(const v8h*)brow;
      ((v8h*)&b[nt])[1] = *(const v8h*)(brow + 8);
    }
#pragma unroll
    for (int nt = 0; nt < 4; ++nt) acc[nt] = wmma_f16(a, b[nt], acc[nt]);
  }

  if (mode == 1) {                       // transposed f16 (V^T): packed b128 stores
    _Float16* outT = (_Float16*)out;
#pragma unroll
    for (int nt = 0; nt < 4; ++nt) {
      const int col = n0 + nt * 16 + lo;
      const float b = bias[col];
      v8h pack;
#pragma unroll
      for (int r = 0; r < 8; ++r) pack[r] = (_Float16)((acc[nt][r] + b) * scale);
      *(v8h*)(outT + (size_t)col * M + m0 + hi * 8) = pack;
    }
  } else if (mode == 0) {                // f16 row-major
    _Float16* o = (_Float16*)out;
#pragma unroll
    for (int nt = 0; nt < 4; ++nt) {
      const int col = n0 + nt * 16 + lo;
      const float b = bias[col];
#pragma unroll
      for (int r = 0; r < 8; ++r) {
        const int row = m0 + r + hi * 8;
        o[(size_t)row * N + col] = (_Float16)((acc[nt][r] + b) * scale);
      }
    }
  } else {                               // f32 row-major (final output)
    float* o = (float*)out;
#pragma unroll
    for (int nt = 0; nt < 4; ++nt) {
      const int col = n0 + nt * 16 + lo;
      const float b = bias[col];
#pragma unroll
      for (int r = 0; r < 8; ++r) {
        const int row = m0 + r + hi * 8;
        o[(size_t)row * N + col] = (acc[nt][r] + b) * scale;
      }
    }
  }
}

// --------------------------------------------------- fused causal attention
// One wave per (head, 16-query tile). Online softmax in exp2 domain.
// Q pre-scaled by 1/sqrt(dk). V supplied transposed [D][S]. K fragments for
// the next chunk and V fragments for this chunk are issued early so the
// softmax VALU stretch overlaps global load latency.
__global__ void __launch_bounds__(128) flash_attn_kernel(
    const _Float16* __restrict__ Qf, const _Float16* __restrict__ Kf,
    const _Float16* __restrict__ Vt, _Float16* __restrict__ Of) {
  __shared__ _Float16 plds[4][16 * 40];   // per-wave P staging, padded stride
  const int lane = threadIdx.x & 31;
  const int wid  = threadIdx.x >> 5;
  const int lo = lane & 15, hi = lane >> 4;
  const int task = blockIdx.x * 4 + wid;
  const int qtiles = SEQ / 16;
  const int h  = task / qtiles;
  const int q0 = (task % qtiles) * 16;
  const size_t hoff = (size_t)h * DK;

  // Q A-fragments for both 32-wide d-chunks of dk=64
  v16h aq[2];
#pragma unroll
  for (int dc = 0; dc < 2; ++dc)
    aq[dc] = load_afrag(Qf + (size_t)(q0 + lo) * D_MODEL + hoff + dc * 32 + hi * 16);

  v8f acc[4] = {};
  float mrow[8], lrow[8];
#pragma unroll
  for (int r = 0; r < 8; ++r) { mrow[r] = -1e30f; lrow[r] = 0.f; }

  const float LOG2E = 1.4426950408889634f;
  const int kend = q0 + 16;               // causal: keys <= q0+15
  _Float16* pl = plds[wid];

  // prime K fragments for the first chunk
  v16h kf[2][2];
#pragma unroll
  for (int st = 0; st < 2; ++st)
#pragma unroll
    for (int dc = 0; dc < 2; ++dc)
      kf[st][dc] = load_bfrag(Kf + (size_t)(st * 16 + lo) * D_MODEL +
                              hoff + dc * 32 + hi * 16);

  for (int k0 = 0; k0 < kend; k0 += 32) {
    // issue V fragments for this chunk early (consumed after softmax)
    v16h vf[4];
#pragma unroll
    for (int nt = 0; nt < 4; ++nt)
      vf[nt] = load_bfrag(Vt + (hoff + nt * 16 + lo) * (size_t)SEQ + k0 + hi * 16);

    // S = Q K^T : two 16x16 score tiles, dk reduced with 2 WMMAs each
    v8f s[2] = {};
#pragma unroll
    for (int st = 0; st < 2; ++st)
#pragma unroll
      for (int dc = 0; dc < 2; ++dc)
        s[st] = wmma_f16(aq[dc], kf[st][dc], s[st]);

    // prefetch next chunk's K fragments (overlaps the softmax VALU work)
    if (k0 + 32 < kend) {
#pragma unroll
      for (int st = 0; st < 2; ++st)
#pragma unroll
        for (int dc = 0; dc < 2; ++dc)
          kf[st][dc] = load_bfrag(Kf + (size_t)(k0 + 32 + st * 16 + lo) * D_MODEL +
                                  hoff + dc * 32 + hi * 16);
    }

    // causal mask + move to log2 domain
#pragma unroll
    for (int st = 0; st < 2; ++st) {
      const int key = k0 + st * 16 + lo;
#pragma unroll
      for (int r = 0; r < 8; ++r) {
        const int row = q0 + r + hi * 8;
        s[st][r] = (key <= row) ? s[st][r] * LOG2E : -1e30f;
      }
    }

    // online softmax update per row (rows live in 16-lane halves)
#pragma unroll
    for (int r = 0; r < 8; ++r) {
      float m = rowmax16(fmaxf(s[0][r], s[1][r]));
      m = fmaxf(m, mrow[r]);
      const float sc = exp2f(mrow[r] - m);
      const float p0 = exp2f(s[0][r] - m);
      const float p1 = exp2f(s[1][r] - m);
      s[0][r] = p0; s[1][r] = p1;
      lrow[r] = lrow[r] * sc + rowsum16(p0 + p1);
      mrow[r] = m;
#pragma unroll
      for (int nt = 0; nt < 4; ++nt) acc[nt][r] *= sc;
    }

    // C-layout P -> LDS -> A-layout fragment (wave-private; ds ops in-order)
#pragma unroll
    for (int st = 0; st < 2; ++st)
#pragma unroll
      for (int r = 0; r < 8; ++r)
        pl[(r + hi * 8) * 40 + st * 16 + lo] = (_Float16)s[st][r];
    asm volatile("s_wait_dscnt 0" ::: "memory");
    v16h pa = load_afrag(pl + lo * 40 + hi * 16);

    // O += P V : V^T rows give contiguous B-fragments (issued at loop top)
#pragma unroll
    for (int nt = 0; nt < 4; ++nt)
      acc[nt] = wmma_f16(pa, vf[nt], acc[nt]);
  }

  // normalize and emit f16 attention output [S][D]
#pragma unroll
  for (int nt = 0; nt < 4; ++nt)
#pragma unroll
    for (int r = 0; r < 8; ++r) {
      const int row = q0 + r + hi * 8;
      const size_t col = hoff + nt * 16 + lo;
      Of[(size_t)row * D_MODEL + col] = (_Float16)(acc[nt][r] / lrow[r]);
    }
}

// ---------------------------------------------------------------- launcher
extern "C" void kernel_launch(void* const* d_in, const int* in_sizes, int n_in,
                              void* d_out, int out_size, void* d_ws, size_t ws_size,
                              hipStream_t stream) {
  (void)in_sizes; (void)n_in; (void)out_size; (void)ws_size;
  const float* x  = (const float*)d_in[0];
  // d_in[1] = causal mask (unused: causality handled analytically)
  const float* Wq = (const float*)d_in[2];
  const float* bq = (const float*)d_in[3];
  const float* Wk = (const float*)d_in[4];
  const float* bk = (const float*)d_in[5];
  const float* Wv = (const float*)d_in[6];
  const float* bv = (const float*)d_in[7];
  const float* Wo = (const float*)d_in[8];
  const float* bo = (const float*)d_in[9];

  // workspace layout (48 MB total)
  char* ws = (char*)d_ws;
  const size_t MB = 1024 * 1024;
  _Float16* xh  = (_Float16*)(ws);            // [S][D]   8 MB
  _Float16* Wqh = (_Float16*)(ws + 8  * MB);  // [D][D]   2 MB each
  _Float16* Wkh = (_Float16*)(ws + 10 * MB);
  _Float16* Wvh = (_Float16*)(ws + 12 * MB);
  _Float16* Woh = (_Float16*)(ws + 14 * MB);
  _Float16* Qh  = (_Float16*)(ws + 16 * MB);  // [S][D]   8 MB (pre-scaled)
  _Float16* Kh  = (_Float16*)(ws + 24 * MB);  // [S][D]   8 MB
  _Float16* Vt  = (_Float16*)(ws + 32 * MB);  // [D][S]   8 MB (transposed)
  _Float16* Ah  = (_Float16*)(ws + 40 * MB);  // [S][D]   8 MB

  const int nx = SEQ * D_MODEL, nw = D_MODEL * D_MODEL;
  cvt_f32_f16_kernel<<<nx / 4 / 256, 256, 0, stream>>>(x,  xh,  nx);
  cvt_f32_f16_kernel<<<nw / 4 / 256, 256, 0, stream>>>(Wq, Wqh, nw);
  cvt_f32_f16_kernel<<<nw / 4 / 256, 256, 0, stream>>>(Wk, Wkh, nw);
  cvt_f32_f16_kernel<<<nw / 4 / 256, 256, 0, stream>>>(Wv, Wvh, nw);
  cvt_f32_f16_kernel<<<nw / 4 / 256, 256, 0, stream>>>(Wo, Woh, nw);

  dim3 g(SEQ / 64, D_MODEL / 64);
  // Q scaled by 1/sqrt(dk)=0.125 here so attention kernel skips the scale
  gemm_wmma_kernel<<<g, 128, 0, stream>>>(xh, Wqh, bq, Qh, SEQ, D_MODEL, D_MODEL, 0.125f, 0);
  gemm_wmma_kernel<<<g, 128, 0, stream>>>(xh, Wkh, bk, Kh, SEQ, D_MODEL, D_MODEL, 1.0f,   0);
  gemm_wmma_kernel<<<g, 128, 0, stream>>>(xh, Wvh, bv, Vt, SEQ, D_MODEL, D_MODEL, 1.0f,   1);

  flash_attn_kernel<<<NUM_HEADS * (SEQ / 16) / 4, 128, 0, stream>>>(Qh, Kh, Vt, Ah);

  gemm_wmma_kernel<<<g, 128, 0, stream>>>(Ah, Woh, bo, d_out, SEQ, D_MODEL, D_MODEL, 1.0f, 2);
}
